// CodeExpander_40364102648037
// MI455X (gfx1250) — compile-verified, matrix operand order checked by reading, not run on hardware
//
#include <hip/hip_runtime.h>
#include <hip/hip_bf16.h>

typedef __attribute__((ext_vector_type(16))) _Float16 v16h;
typedef __attribute__((ext_vector_type(8)))  _Float16 v8h;
typedef __attribute__((ext_vector_type(8)))  float    v8f;

#define DMODEL 512
#define NHEAD  8
#define DHEAD  64

// ---------------------------------------------------------------------------
// WMMA fragment loader (CDNA5 16-bit A/B layout, ISA 7.12.2).
// tile must be [idx][k] with k contiguous (A: idx=row m; B: idx=col n, i.e.
// the tile is stored K-major).  Per lane: two 16B K-contiguous runs at
// k = koff + 8*(lane/16) and +16  ->  two ds_load_b128.
// ---------------------------------------------------------------------------
__device__ __forceinline__ v16h load_frag_km(const _Float16* tile, int ld, int koff) {
  const int lane = threadIdx.x & 31;
  const int m = lane & 15;
  const int h8 = (lane >> 4) << 3;
  const _Float16* p = tile + m * ld + koff + h8;
  union { v16h v; v8h h[2]; } u;
  u.h[0] = *(const v8h*)(p);
  u.h[1] = *(const v8h*)(p + 16);
  return u.v;
}

__device__ __forceinline__ v8f wmma_f16(v16h a, v16h b, v8f c) {
  return __builtin_amdgcn_wmma_f32_16x16x32_f16(false, a, false, b, (short)0, c,
                                                false, false);
}

__device__ __forceinline__ v8h cvt_v8h(float4 a, float4 b) {
  v8h h;
  h[0] = (_Float16)a.x; h[1] = (_Float16)a.y; h[2] = (_Float16)a.z; h[3] = (_Float16)a.w;
  h[4] = (_Float16)b.x; h[5] = (_Float16)b.y; h[6] = (_Float16)b.z; h[7] = (_Float16)b.w;
  return h;
}

__device__ __forceinline__ float half16_max(float v) {
  v = fmaxf(v, __shfl_xor(v, 1, 32));
  v = fmaxf(v, __shfl_xor(v, 2, 32));
  v = fmaxf(v, __shfl_xor(v, 4, 32));
  v = fmaxf(v, __shfl_xor(v, 8, 32));
  return v;
}
__device__ __forceinline__ float half16_sum(float v) {
  v += __shfl_xor(v, 1, 32);
  v += __shfl_xor(v, 2, 32);
  v += __shfl_xor(v, 4, 32);
  v += __shfl_xor(v, 8, 32);
  return v;
}

// ---------------------------------------------------------------------------
// GEMM: out[M,N] = epi(A[M,K] @ W[K,N] + bias[N])
// epi 0: none; 1: += res; 2: out = silu(aux)*val
// Block tile 128x128, 8 waves, wave tile 32x64 -> 8 WMMA per K-step per wave
// from 2 A-frags + 4 B-frags (B reused across both A rows).
// N,K template constants -> immediate-offset strided loads; LDS double-
// buffered (stage k+1 while computing k, one barrier per K-step).
// ---------------------------------------------------------------------------
#define BM 128
#define BN 128
#define BK 32

template <int N, int K>
__global__ __launch_bounds__(256) void gemm_wmma_kernel(
    const float* __restrict__ A, const float* __restrict__ W,
    const float* __restrict__ bias, const float* __restrict__ res,
    float* __restrict__ out, const float* __restrict__ aux, int epi) {
  __shared__ _Float16 As[2][BM][BK];   // row-major (k contiguous)
  __shared__ _Float16 Wt[2][BN][BK];   // K-major  (k contiguous per out col)

  const int tid = threadIdx.x;
  const int wave = tid >> 5;
  const int lane = tid & 31;
  const int m0 = blockIdx.y * BM;
  const int n0 = blockIdx.x * BN;
  const int rt = (wave & 3) * 2;   // row subtiles rt, rt+1   (of 8)
  const int ct = (wave >> 2) * 4;  // col subtiles ct..ct+3   (of 8)

  // staging coordinates (chunks of 8 along k)
  const int sr = tid >> 2;            // 0..63 (+64 on 2nd pass)
  const int sc8 = (tid & 3) * 8;      // k chunk

  auto stage = [&](int k0, int buf) {
#pragma unroll
    for (int t = 0; t < 2; ++t) {
      int r = sr + t * 64;
      const float4* g = (const float4*)&A[(size_t)(m0 + r) * K + k0 + sc8];
      *(v8h*)&As[buf][r][sc8] = cvt_v8h(g[0], g[1]);
    }
#pragma unroll
    for (int t = 0; t < 2; ++t) {
      int n = sr + t * 64;
      const float* wp = &W[(size_t)(k0 + sc8) * N + n0 + n];
      v8h h;
#pragma unroll
      for (int i = 0; i < 8; ++i) h[i] = (_Float16)wp[i * N];  // imm offsets
      *(v8h*)&Wt[buf][n][sc8] = h;
    }
  };

  v8f acc[2][4] = {};
  constexpr int NK = K / BK;

  stage(0, 0);
  __syncthreads();
#pragma unroll 1
  for (int ks = 0; ks < NK; ++ks) {
    const int buf = ks & 1;
    if (ks + 1 < NK) stage((ks + 1) * BK, buf ^ 1);
    if (ks + 2 < NK) {  // global_prefetch of tile after next
      __builtin_prefetch(&A[(size_t)(m0 + sr) * K + (ks + 2) * BK + sc8], 0, 0);
      __builtin_prefetch(&W[(size_t)((ks + 2) * BK + sc8) * N + n0 + sr], 0, 0);
    }
    v16h a0 = load_frag_km(&As[buf][(rt + 0) * 16][0], BK, 0);
    v16h a1 = load_frag_km(&As[buf][(rt + 1) * 16][0], BK, 0);
#pragma unroll
    for (int cc = 0; cc < 4; ++cc) {
      v16h bf = load_frag_km(&Wt[buf][(ct + cc) * 16][0], BK, 0);
      acc[0][cc] = wmma_f16(a0, bf, acc[0][cc]);
      acc[1][cc] = wmma_f16(a1, bf, acc[1][cc]);
    }
    __syncthreads();
  }

  const int ncol = lane & 15;
  const int rowh = (lane >> 4) << 3;
#pragma unroll
  for (int rr = 0; rr < 2; ++rr) {
#pragma unroll
    for (int cc = 0; cc < 4; ++cc) {
      const v8f& a = acc[rr][cc];
      int gn = n0 + (ct + cc) * 16 + ncol;
      float bv = bias[gn];
#pragma unroll
      for (int v = 0; v < 8; ++v) {
        int gm = m0 + (rt + rr) * 16 + rowh + v;
        size_t o = (size_t)gm * N + gn;
        float val = a[v] + bv;
        if (epi == 1) val += res[o];
        else if (epi == 2) { float g = aux[o]; val *= g / (1.f + __expf(-g)); }
        out[o] = val;
      }
    }
  }
}

// ---------------------------------------------------------------------------
// Fused flash attention.  Buffers [B*L, DMODEL] token-major, head hh at cols
// hh*64.  grid = (Lq/128, NHEAD, B); 8 waves; wave owns one 16-row Q tile.
// K tile row-major (feat contiguous -> score B frags are b128 loads);
// V tile feature-major (key contiguous -> PV B frags are b128 loads).
// K/V tiles double-buffered: one barrier per 32-key step.
// ---------------------------------------------------------------------------
__global__ __launch_bounds__(256) void attn_wmma_kernel(
    const float* __restrict__ Q, const float* __restrict__ K,
    const float* __restrict__ V, float* __restrict__ O,
    int Lq, int Lk, int causal) {
  __shared__ _Float16 Ks[2][32][64];    // [key][feat]
  __shared__ _Float16 Vst[2][64][32];   // [feat][key]
  __shared__ _Float16 Qs[8][16][64];    // per-wave [row][feat]
  __shared__ _Float16 Ps[8][16][32];    // per-wave [row][key]

  const int tid = threadIdx.x;
  const int wave = tid >> 5;
  const int lane = tid & 31;
  const int b = blockIdx.z;
  const int hh = blockIdx.y;
  const int q0 = blockIdx.x * 128 + wave * 16;
  const size_t qbase = (size_t)b * Lq;
  const size_t kbase = (size_t)b * Lk;
  const int c0 = hh * DHEAD;
  const float scale = 0.125f;  // 1/sqrt(64)

  // staging coords
  const int kr = tid >> 3, kc8 = (tid & 7) * 8;  // K: [key][feat] chunks
  const int vf = tid >> 2, vkc = (tid & 3) * 8;  // V: [feat][key] chunks

  auto stageKV = [&](int kt0, int buf) {
    const float4* g = (const float4*)&K[(kbase + kt0 + kr) * DMODEL + c0 + kc8];
    *(v8h*)&Ks[buf][kr][kc8] = cvt_v8h(g[0], g[1]);
    const float* vp = &V[(kbase + kt0 + vkc) * DMODEL + c0 + vf];
    v8h h;
#pragma unroll
    for (int i = 0; i < 8; ++i) h[i] = (_Float16)vp[i * DMODEL];  // imm offsets
    *(v8h*)&Vst[buf][vf][vkc] = h;
  };

  // per-wave Q tile (pre-scaled): 16x64 = 128 v8h chunks, 4 per lane
#pragma unroll
  for (int t = 0; t < 4; ++t) {
    int id = lane + 32 * t;
    int r = id >> 3, c8 = (id & 7) * 8;
    const float4* g = (const float4*)&Q[(qbase + q0 + r) * DMODEL + c0 + c8];
    float4 f0 = g[0], f1 = g[1];
    f0.x *= scale; f0.y *= scale; f0.z *= scale; f0.w *= scale;
    f1.x *= scale; f1.y *= scale; f1.z *= scale; f1.w *= scale;
    *(v8h*)&Qs[wave][r][c8] = cvt_v8h(f0, f1);
  }
  // Q frags are loop-invariant: order the per-wave LDS RAW, then hoist.
  __builtin_amdgcn_wave_barrier();
  asm volatile("s_wait_dscnt 0" ::: "memory");
  const v16h aq0 = load_frag_km(&Qs[wave][0][0], 64, 0);
  const v16h aq1 = load_frag_km(&Qs[wave][0][0], 64, 32);

  v8f o0 = {}, o1 = {}, o2 = {}, o3 = {};
  float mrow[8], lrow[8];
#pragma unroll
  for (int v = 0; v < 8; ++v) { mrow[v] = -1e30f; lrow[v] = 0.f; }

  const int ncol = lane & 15;
  const int rowh = (lane >> 4) << 3;

  stageKV(0, 0);
  __syncthreads();
#pragma unroll 1
  for (int kt0 = 0; kt0 < Lk; kt0 += 32) {
    const int buf = (kt0 >> 5) & 1;
    if (kt0 + 32 < Lk) stageKV(kt0 + 32, buf ^ 1);

    // S[16x32] = Q(16x64) @ K^T(64x32keys)
    v8f s0 = {}, s1 = {};
    s0 = wmma_f16(aq0, load_frag_km(&Ks[buf][0][0], 64, 0), s0);
    s0 = wmma_f16(aq1, load_frag_km(&Ks[buf][0][0], 64, 32), s0);
    s1 = wmma_f16(aq0, load_frag_km(&Ks[buf][16][0], 64, 0), s1);
    s1 = wmma_f16(aq1, load_frag_km(&Ks[buf][16][0], 64, 32), s1);

    const int key0 = kt0 + ncol;
    const int key1 = key0 + 16;
    if (causal) {
#pragma unroll
      for (int v = 0; v < 8; ++v) {
        int row = q0 + rowh + v;
        if (key0 > row) s0[v] = -1e30f;
        if (key1 > row) s1[v] = -1e30f;
      }
    }

    // online softmax (rows live in 16-lane halves)
#pragma unroll
    for (int v = 0; v < 8; ++v) {
      float tmax = half16_max(fmaxf(s0[v], s1[v]));
      float mnew = fmaxf(mrow[v], tmax);
      float corr = __expf(mrow[v] - mnew);
      mrow[v] = mnew;
      float p0 = __expf(s0[v] - mnew);
      float p1 = __expf(s1[v] - mnew);
      lrow[v] = lrow[v] * corr + half16_sum(p0 + p1);
      o0[v] *= corr; o1[v] *= corr; o2[v] *= corr; o3[v] *= corr;
      int prow = rowh + v;
      Ps[wave][prow][ncol] = (_Float16)p0;
      Ps[wave][prow][ncol + 16] = (_Float16)p1;
    }

    // per-wave LDS RAW: LDS in-order per wave; fence compiler + drain DScnt
    __builtin_amdgcn_wave_barrier();
    asm volatile("s_wait_dscnt 0" ::: "memory");

    // O(16x64) += P(16x32) @ V(32x64)
    v16h ap = load_frag_km(&Ps[wave][0][0], 32, 0);
    o0 = wmma_f16(ap, load_frag_km(&Vst[buf][0][0], 32, 0), o0);
    o1 = wmma_f16(ap, load_frag_km(&Vst[buf][16][0], 32, 0), o1);
    o2 = wmma_f16(ap, load_frag_km(&Vst[buf][32][0], 32, 0), o2);
    o3 = wmma_f16(ap, load_frag_km(&Vst[buf][48][0], 32, 0), o3);
    __syncthreads();
  }

#pragma unroll
  for (int v = 0; v < 8; ++v) {
    int gm = q0 + rowh + v;
    float rinv = 1.f / lrow[v];
    size_t base = (qbase + gm) * DMODEL + c0;
    O[base + 0  + ncol] = o0[v] * rinv;
    O[base + 16 + ncol] = o1[v] * rinv;
    O[base + 32 + ncol] = o2[v] * rinv;
    O[base + 48 + ncol] = o3[v] * rinv;
  }
}

// ---------------------------------------------------------------------------
// Elementwise kernels
// ---------------------------------------------------------------------------
__global__ __launch_bounds__(128) void embed_kernel(
    const int* __restrict__ codes, const float* __restrict__ emb,
    float* __restrict__ out) {
  int t = blockIdx.x;
  int c = codes[t];
  const float4* src = (const float4*)(emb + (size_t)c * DMODEL);
  float4* dst = (float4*)(out + (size_t)t * DMODEL);
  dst[threadIdx.x] = src[threadIdx.x];
}

__global__ __launch_bounds__(128) void embed_dec_kernel(
    const int* __restrict__ codes, const float* __restrict__ emb,
    float* __restrict__ out, int L) {
  int t = blockIdx.x;
  int b = t / L, pos = t - b * L;
  int id = (pos == 0) ? 1 : codes[b * L + pos - 1];  // EOS_ID = 1 shift
  const float4* src = (const float4*)(emb + (size_t)id * DMODEL);
  float4* dst = (float4*)(out + (size_t)t * DMODEL);
  dst[threadIdx.x] = src[threadIdx.x];
}

__global__ __launch_bounds__(256) void rmsnorm_kernel(
    const float* __restrict__ x, const float* __restrict__ w,
    float* __restrict__ out) {
  __shared__ float red[8];
  int row = blockIdx.x, tid = threadIdx.x;
  const float2* xr = (const float2*)(x + (size_t)row * DMODEL);
  float2 a = xr[tid];
  float s = a.x * a.x + a.y * a.y;
#pragma unroll
  for (int m = 1; m < 32; m <<= 1) s += __shfl_xor(s, m, 32);
  if ((tid & 31) == 0) red[tid >> 5] = s;
  __syncthreads();
  float tot = 0.f;
#pragma unroll
  for (int i = 0; i < 8; ++i) tot += red[i];
  float rinv = rsqrtf(tot / (float)DMODEL + 1e-6f);
  float2 wv = ((const float2*)w)[tid];
  float2 r;
  r.x = a.x * rinv * wv.x;
  r.y = a.y * rinv * wv.y;
  ((float2*)(out + (size_t)row * DMODEL))[tid] = r;
}

__global__ __launch_bounds__(256) void rope_kernel(float* __restrict__ x, int L) {
  int tok = blockIdx.x;
  int pos = tok % L;
  int hh = threadIdx.x >> 5, i = threadIdx.x & 31;
  size_t base = (size_t)tok * DMODEL + hh * DHEAD;
  float x1 = x[base + i], x2 = x[base + 32 + i];
  float inv = __powf(10000.f, -(float)i / 32.f);
  float ang = (float)pos * inv;
  float c = __cosf(ang), s = __sinf(ang);
  x[base + i] = x1 * c - x2 * s;
  x[base + 32 + i] = x1 * s + x2 * c;
}

// ---------------------------------------------------------------------------
// Host orchestration
// ---------------------------------------------------------------------------
namespace {
struct EncB {
  const float *n1, *n2;
  const float *qw, *qb, *kw, *kb, *vw, *vb, *ow, *ob;
  const float *w1, *b1, *w2, *b2, *w3, *b3;
};
struct DecB {
  const float *n1, *n2, *n3;
  const float *sqw, *sqb, *skw, *skb, *svw, *svb, *sow, *sob;
  const float *cqw, *cqb, *ckw, *ckb, *cvw, *cvb, *cow, *cob;
  const float *w1, *b1, *w2, *b2, *w3, *b3;
};
}  // namespace

extern "C" void kernel_launch(void* const* d_in, const int* in_sizes, int n_in,
                              void* d_out, int out_size, void* d_ws, size_t ws_size,
                              hipStream_t stream) {
  (void)in_sizes; (void)out_size; (void)ws_size;
  const int D = DMODEL, FFN = 2048, B = 4, Le = 1024, Ld = 2048;
  const int Me = B * Le, Md = B * Ld;
  if (n_in < 172) return;

  const int* codes_hi = (const int*)d_in[0];
  const int* codes_lo = (const int*)d_in[1];
  const float* emb_hi = (const float*)d_in[2];
  const float* emb_lo = (const float*)d_in[3];

  int p = 4;
  auto F = [&](int i) { return (const float*)d_in[i]; };
  EncB enc[4];
  for (int i = 0; i < 4; ++i) {
    enc[i].n1 = F(p++); enc[i].n2 = F(p++);
    enc[i].qw = F(p++); enc[i].qb = F(p++); enc[i].kw = F(p++); enc[i].kb = F(p++);
    enc[i].vw = F(p++); enc[i].vb = F(p++); enc[i].ow = F(p++); enc[i].ob = F(p++);
    enc[i].w1 = F(p++); enc[i].b1 = F(p++); enc[i].w2 = F(p++); enc[i].b2 = F(p++);
    enc[i].w3 = F(p++); enc[i].b3 = F(p++);
  }
  const float* enc_fn = F(p++);
  DecB dec[4];
  for (int i = 0; i < 4; ++i) {
    dec[i].n1 = F(p++); dec[i].n2 = F(p++); dec[i].n3 = F(p++);
    dec[i].sqw = F(p++); dec[i].sqb = F(p++); dec[i].skw = F(p++); dec[i].skb = F(p++);
    dec[i].svw = F(p++); dec[i].svb = F(p++); dec[i].sow = F(p++); dec[i].sob = F(p++);
    dec[i].cqw = F(p++); dec[i].cqb = F(p++); dec[i].ckw = F(p++); dec[i].ckb = F(p++);
    dec[i].cvw = F(p++); dec[i].cvb = F(p++); dec[i].cow = F(p++); dec[i].cob = F(p++);
    dec[i].w1 = F(p++); dec[i].b1 = F(p++); dec[i].w2 = F(p++); dec[i].b2 = F(p++);
    dec[i].w3 = F(p++); dec[i].b3 = F(p++);
  }
  const float* dec_fn = F(p++);
  const float* out_w = F(p++);
  const float* out_b = F(p++);

  // Workspace layout (fp32)
  float* bufX = (float*)d_ws;                       // Md*D running stream
  float* bufH = bufX + (size_t)Md * D;              // Md*D normed
  float* bufQ = bufH + (size_t)Md * D;
  float* bufK = bufQ + (size_t)Md * D;
  float* bufV = bufK + (size_t)Md * D;
  float* bufA = bufV + (size_t)Md * D;              // attn out
  float* bufM = bufA + (size_t)Md * D;              // Me*D encoder memory
  float* bufG = bufM + (size_t)Me * D;              // Md*FFN

  auto gemm = [&](const float* A, const float* W, const float* bias,
                  const float* res, float* out, const float* aux,
                  int M, int N, int K, int epi) {
    dim3 grid(N / BN, M / BM);
    if (N == 512 && K == 512)
      gemm_wmma_kernel<512, 512><<<grid, 256, 0, stream>>>(A, W, bias, res, out, aux, epi);
    else if (N == 2048 && K == 512)
      gemm_wmma_kernel<2048, 512><<<grid, 256, 0, stream>>>(A, W, bias, res, out, aux, epi);
    else if (N == 512 && K == 2048)
      gemm_wmma_kernel<512, 2048><<<grid, 256, 0, stream>>>(A, W, bias, res, out, aux, epi);
    else if (N == 256 && K == 512)
      gemm_wmma_kernel<256, 512><<<grid, 256, 0, stream>>>(A, W, bias, res, out, aux, epi);
  };
  auto rms = [&](const float* x, const float* w, float* out, int M) {
    rmsnorm_kernel<<<M, 256, 0, stream>>>(x, w, out);
  };

  // ---------------- Encoder ----------------
  embed_kernel<<<Me, 128, 0, stream>>>(codes_hi, emb_hi, bufX);
  for (int i = 0; i < 4; ++i) {
    const EncB& e = enc[i];
    rms(bufX, e.n1, bufH, Me);
    gemm(bufH, e.qw, e.qb, nullptr, bufQ, nullptr, Me, D, D, 0);
    gemm(bufH, e.kw, e.kb, nullptr, bufK, nullptr, Me, D, D, 0);
    gemm(bufH, e.vw, e.vb, nullptr, bufV, nullptr, Me, D, D, 0);
    rope_kernel<<<Me, 256, 0, stream>>>(bufQ, Le);
    rope_kernel<<<Me, 256, 0, stream>>>(bufK, Le);
    attn_wmma_kernel<<<dim3(Le / 128, NHEAD, B), 256, 0, stream>>>(
        bufQ, bufK, bufV, bufA, Le, Le, 0);
    gemm(bufA, e.ow, e.ob, bufX, bufX, nullptr, Me, D, D, 1);
    rms(bufX, e.n2, bufH, Me);
    gemm(bufH, e.w1, e.b1, nullptr, bufG, nullptr, Me, FFN, D, 0);
    gemm(bufH, e.w2, e.b2, nullptr, bufG, bufG, Me, FFN, D, 2);
    gemm(bufG, e.w3, e.b3, bufX, bufX, nullptr, Me, D, FFN, 1);
  }
  rms(bufX, enc_fn, bufM, Me);

  // ---------------- Decoder ----------------
  embed_dec_kernel<<<Md, 128, 0, stream>>>(codes_lo, emb_lo, bufX, Ld);
  for (int i = 0; i < 4; ++i) {
    const DecB& d = dec[i];
    // causal self-attention
    rms(bufX, d.n1, bufH, Md);
    gemm(bufH, d.sqw, d.sqb, nullptr, bufQ, nullptr, Md, D, D, 0);
    gemm(bufH, d.skw, d.skb, nullptr, bufK, nullptr, Md, D, D, 0);
    gemm(bufH, d.svw, d.svb, nullptr, bufV, nullptr, Md, D, D, 0);
    rope_kernel<<<Md, 256, 0, stream>>>(bufQ, Ld);
    rope_kernel<<<Md, 256, 0, stream>>>(bufK, Ld);
    attn_wmma_kernel<<<dim3(Ld / 128, NHEAD, B), 256, 0, stream>>>(
        bufQ, bufK, bufV, bufA, Ld, Ld, 1);
    gemm(bufA, d.sow, d.sob, bufX, bufX, nullptr, Md, D, D, 1);
    // cross-attention against encoder memory
    rms(bufX, d.n2, bufH, Md);
    gemm(bufH, d.cqw, d.cqb, nullptr, bufQ, nullptr, Md, D, D, 0);
    gemm(bufM, d.ckw, d.ckb, nullptr, bufK, nullptr, Me, D, D, 0);
    gemm(bufM, d.cvw, d.cvb, nullptr, bufV, nullptr, Me, D, D, 0);
    rope_kernel<<<Md, 256, 0, stream>>>(bufQ, Ld);
    rope_kernel<<<Me, 256, 0, stream>>>(bufK, Le);
    attn_wmma_kernel<<<dim3(Ld / 128, NHEAD, B), 256, 0, stream>>>(
        bufQ, bufK, bufV, bufA, Ld, Le, 0);
    gemm(bufA, d.cow, d.cob, bufX, bufX, nullptr, Md, D, D, 1);
    // FFN
    rms(bufX, d.n3, bufH, Md);
    gemm(bufH, d.w1, d.b1, nullptr, bufG, nullptr, Md, FFN, D, 0);
    gemm(bufH, d.w2, d.b2, nullptr, bufG, bufG, Md, FFN, D, 2);
    gemm(bufG, d.w3, d.b3, bufX, bufX, nullptr, Md, D, FFN, 1);
  }
  rms(bufX, dec_fn, bufH, Md);
  gemm(bufH, out_w, out_b, nullptr, (float*)d_out, nullptr, Md, 256, D, 0);
}
